// CrossAttention_L2P_82016695485239
// MI455X (gfx1250) — compile-verified, matrix operand order checked by reading, not run on hardware
//
#include <hip/hip_runtime.h>
#include <hip/hip_bf16.h>
#include <math.h>

// ---------------------------------------------------------------------------
// Problem constants (from reference)
// ---------------------------------------------------------------------------
constexpr int kB   = 8;     // batch
constexpr int kC   = 256;   // patch channels
constexpr int kN   = 4096;  // D*H*W patches
constexpr int kM   = 512;   // mol feature dim
constexpr int kE   = 16;    // mol tokens
constexpr int kHID = 512;   // hidden
constexpr int kNH  = 8;     // heads
constexpr int kHD  = 64;    // head dim
constexpr float kEPS = 1.1920929e-07f;

typedef __attribute__((ext_vector_type(16))) __bf16 v16bf;
typedef __attribute__((ext_vector_type(8)))  __bf16 v8bf;
typedef __attribute__((ext_vector_type(8)))  float  v8f;
typedef __attribute__((ext_vector_type(4)))  unsigned int u32x4;
typedef __attribute__((ext_vector_type(8)))  int i32x8;
typedef __attribute__((ext_vector_type(4)))  int i32x4;

__device__ __forceinline__ float gelu_erf(float x) {
    return 0.5f * x * (1.0f + erff(x * 0.70710678118f));
}

// Flat LDS pointer -> byte offset within LDS (flat addr low 32 bits = offset).
__device__ __forceinline__ unsigned lds_byte_offset(const void* p) {
    return (unsigned)(unsigned long long)p;
}

// ---------------------------------------------------------------------------
// 1) Positional encoding MLP: [N,3] -> gelu(@pe_w1+b1) -> @pe_w2+b2 -> [N,512]
// ---------------------------------------------------------------------------
__global__ void pos_enc_kernel(const float* __restrict__ pe_w1, const float* __restrict__ pe_b1,
                               const float* __restrict__ pe_w2, const float* __restrict__ pe_b2,
                               float* __restrict__ pos_enc) {
    __shared__ float h1[256];
    int n = blockIdx.x;        // 0..4095
    int t = threadIdx.x;       // 0..255
    float cd = (float)(n >> 8)        * (1.0f / 16.0f);
    float ch = (float)((n >> 4) & 15) * (1.0f / 16.0f);
    float cw = (float)(n & 15)        * (1.0f / 16.0f);
    float a = cd * pe_w1[t] + ch * pe_w1[256 + t] + cw * pe_w1[512 + t] + pe_b1[t];
    h1[t] = gelu_erf(a);
    __syncthreads();
    #pragma unroll
    for (int rep = 0; rep < 2; ++rep) {
        int j = t + rep * 256;
        float acc = pe_b2[j];
        for (int i = 0; i < 256; ++i) acc += h1[i] * pe_w2[i * kHID + j];
        pos_enc[(size_t)n * kHID + j] = acc;
    }
}

// ---------------------------------------------------------------------------
// 2) wkv fp32 [256,1024] -> bf16 transposed [1024,256] (B-fragment friendly)
// ---------------------------------------------------------------------------
__global__ void wkv_convert_kernel(const float* __restrict__ wkv, __bf16* __restrict__ wkv_t) {
    int j = blockIdx.x;   // 0..1023
    int c = threadIdx.x;  // 0..255
    wkv_t[(size_t)j * kC + c] = (__bf16)wkv[(size_t)c * 1024 + j];
}

// ---------------------------------------------------------------------------
// 3) patch RMSNorm over C + transpose [B,C,N] -> bf16 [B,N,C]
//    Two passes over C; 33.5MB input is L2-resident (192MB L2).
// ---------------------------------------------------------------------------
__global__ void patch_norm_kernel(const float* __restrict__ patch,
                                  const float* __restrict__ norm_patch_w,
                                  __bf16* __restrict__ patchN) {
    int b = blockIdx.x >> 4;                        // 8 batches
    int n = ((blockIdx.x & 15) << 8) + threadIdx.x; // 16 tiles * 256 n
    const float* base = patch + ((size_t)b * kC) * kN + n;
    float ss = 0.f;
    for (int c = 0; c < kC; ++c) { float x = base[(size_t)c * kN]; ss += x * x; }
    float r = rsqrtf(ss * (1.0f / kC) + kEPS);
    __bf16* outb = patchN + ((size_t)b * kN + n) * kC;
    for (int c0 = 0; c0 < kC; c0 += 8) {
        v8bf o;
        #pragma unroll
        for (int i = 0; i < 8; ++i)
            o[i] = (__bf16)(base[(size_t)(c0 + i) * kN] * r * norm_patch_w[c0 + i]);
        *(v8bf*)(outb + c0) = o;
    }
}

// ---------------------------------------------------------------------------
// 4) kv GEMM: [B*N,256]bf16 @ [256,1024]bf16 -> +bias +pos_enc ->
//    k bf16 [bh][n][d], v^T bf16 [bh][d][n].
//    A tile (16x256 bf16 = 8KB) is brought in by the Tensor Data Mover:
//    2-D D# with data_size=8B, row=64 units, 16 rows, contiguous stride.
// ---------------------------------------------------------------------------
__global__ void kv_wmma_kernel(const __bf16* __restrict__ patchN,
                               const __bf16* __restrict__ wkv_t,
                               const float* __restrict__ bkv,
                               const float* __restrict__ pos_enc,
                               __bf16* __restrict__ kmat,
                               __bf16* __restrict__ vt) {
    __shared__ __attribute__((aligned(32))) __bf16 Atile[16 * kC];  // 8 KB
    int rowTile = blockIdx.x;       // tile of 16 rows of flattened [B*N]
    int half    = blockIdx.y;       // 0: k cols, 1: v cols
    int t = threadIdx.x, lane = t & 31, wave = t >> 5;

    // --- TDM load of the A tile (issued once, from wave 0) ---
    if (wave == 0) {
        unsigned long long ga =
            (unsigned long long)(const void*)(patchN + (size_t)rowTile * 16 * kC);
        u32x4 g0 = {
            1u,                                   // count=1, user mode
            lds_byte_offset(Atile),               // lds_addr
            (unsigned)ga,                         // global_addr[31:0]
            ((unsigned)(ga >> 32) & 0x01FFFFFFu)  // global_addr[56:32]
                | (2u << 30)                      // type = 2 ("image")
        };
        i32x8 g1 = {
            (int)(3u << 16),          // data_size = 8B, workgroup_mask = 0
            (int)(64u << 16),         // tensor_dim0[15:0] = 64 units/row
            (int)(0x8000u << 16),     // tensor_dim1[15:0] = 32768 rows
            (int)(64u << 16),         // tensor_dim1[31:16]=0 | tile_dim0 = 64
            16,                       // tile_dim1 = 16 rows (tile_dim2 = 0)
            64,                       // tensor_dim0_stride = 64 units
            0, 0
        };
        i32x4 gz4 = { 0, 0, 0, 0 };
        i32x8 gz8 = { 0, 0, 0, 0, 0, 0, 0, 0 };
        __builtin_amdgcn_tensor_load_to_lds(g0, g1, gz4, gz4, gz8, 0);
    }
    __builtin_amdgcn_s_wait_tensorcnt(0);
    __syncthreads();

    int jbase = half * kHID + wave * 64;
    int arow  = lane & 15;
    int khalf = (lane >> 4) << 4;   // 0 or 16: K-half per lane group
    v8f acc[4] = {};
    for (int ks = 0; ks < kC; ks += 32) {
        v16bf a = *(const v16bf*)&Atile[arow * kC + ks + khalf];
        #pragma unroll
        for (int tj = 0; tj < 4; ++tj) {
            int j = jbase + tj * 16 + (lane & 15);
            v16bf bf = *(const v16bf*)(wkv_t + (size_t)j * kC + ks + khalf);
            acc[tj] = __builtin_amdgcn_wmma_f32_16x16x32_bf16(
                false, a, false, bf, (short)0, acc[tj], false, false);
        }
    }

    size_t nglob0 = (size_t)rowTile * 16;       // one batch per tile (16 | 4096)
    size_t gi = nglob0 + ((size_t)(lane >> 4) << 3);
    size_t b  = gi >> 12;
    size_t n0 = gi & (kN - 1);
    #pragma unroll
    for (int tj = 0; tj < 4; ++tj) {
        int j  = jbase + tj * 16 + (lane & 15);
        int jj = j & (kHID - 1);                // within-half column
        int h = jj >> 6, dd = jj & 63;
        float bias = bkv[j];
        if (half == 0) {
            #pragma unroll
            for (int r = 0; r < 8; ++r) {
                float val = acc[tj][r] + bias + pos_enc[(n0 + r) * kHID + jj];
                kmat[(((b * kNH + h) * kN) + n0 + r) * kHD + dd] = (__bf16)val;
            }
        } else {
            v8bf o;   // 8 consecutive n per lane -> single 16B store
            #pragma unroll
            for (int r = 0; r < 8; ++r)
                o[r] = (__bf16)(acc[tj][r] + bias + pos_enc[(n0 + r) * kHID + jj]);
            *(v8bf*)(vt + (((b * kNH + h) * kHD) + dd) * kN + n0) = o;
        }
    }
}

// ---------------------------------------------------------------------------
// 5) mol RMSNorm (kept fp32 for residual) + q projection (scaled by d^-1/2),
//    q stored bf16 head-major [bh][e][d]. One block per batch.
// ---------------------------------------------------------------------------
__global__ void q_mol_kernel(const float* __restrict__ mol,
                             const float* __restrict__ norm_mol_w,
                             const float* __restrict__ wq, const float* __restrict__ bq,
                             float* __restrict__ mol_norm, __bf16* __restrict__ qb) {
    __shared__ float mn[kE * kM];  // 32 KB
    __shared__ float red[256];
    int b = blockIdx.x, t = threadIdx.x;
    const float* molb = mol + (size_t)b * kE * kM;
    for (int e = 0; e < kE; ++e) {
        float s = 0.f;
        for (int c = t; c < kM; c += 256) { float x = molb[e * kM + c]; s += x * x; }
        red[t] = s; __syncthreads();
        for (int off = 128; off > 0; off >>= 1) {
            if (t < off) red[t] += red[t + off];
            __syncthreads();
        }
        float r = rsqrtf(red[0] * (1.0f / kM) + kEPS);
        __syncthreads();
        for (int c = t; c < kM; c += 256) {
            float v = molb[e * kM + c] * r * norm_mol_w[c];
            mn[e * kM + c] = v;
            mol_norm[(size_t)b * kE * kM + e * kM + c] = v;
        }
        __syncthreads();
    }
    const float scale = 0.125f;  // 64^-0.5
    for (int o = t; o < kE * kHID; o += 256) {
        int e = o >> 9, j = o & (kHID - 1);
        float acc = bq[j];
        for (int m = 0; m < kM; ++m) acc += mn[e * kM + m] * wq[(size_t)m * kHID + j];
        int h = j >> 6, dd = j & 63;
        qb[(((size_t)b * kNH + h) * kE + e) * kHD + dd] = (__bf16)(acc * scale);
    }
}

// ---------------------------------------------------------------------------
// 6) scores = q @ k^T per (b,h): [16,64] x [64,N] -> fp32 [bh][e][n]
// ---------------------------------------------------------------------------
__global__ void scores_wmma_kernel(const __bf16* __restrict__ qb,
                                   const __bf16* __restrict__ kmat,
                                   float* __restrict__ scores) {
    int bh = blockIdx.x;
    int t = threadIdx.x, lane = t & 31, wave = t >> 5;
    int n0 = blockIdx.y * 512 + wave * 64;
    int khalf = (lane >> 4) << 4;
    v8f acc[4] = {};
    #pragma unroll
    for (int ks = 0; ks < kHD; ks += 32) {
        v16bf a = *(const v16bf*)(qb + ((size_t)bh * kE + (lane & 15)) * kHD + ks + khalf);
        #pragma unroll
        for (int tn = 0; tn < 4; ++tn) {
            int n = n0 + tn * 16 + (lane & 15);
            v16bf bf = *(const v16bf*)(kmat + ((size_t)bh * kN + n) * kHD + ks + khalf);
            acc[tn] = __builtin_amdgcn_wmma_f32_16x16x32_bf16(
                false, a, false, bf, (short)0, acc[tn], false, false);
        }
    }
    #pragma unroll
    for (int tn = 0; tn < 4; ++tn) {
        int n = n0 + tn * 16 + (lane & 15);
        #pragma unroll
        for (int r = 0; r < 8; ++r) {
            int e = r + ((lane >> 4) << 3);
            scores[((size_t)bh * kE + e) * kN + n] = acc[tn][r];
        }
    }
}

// ---------------------------------------------------------------------------
// 7) row softmax over N=4096, fp32 in -> bf16 probs out. One block per row;
//    each thread owns 16 contiguous columns (vectorized 2x16B store).
// ---------------------------------------------------------------------------
__global__ void softmax_kernel(const float* __restrict__ scores, __bf16* __restrict__ probs) {
    __shared__ float red[256];
    int row = blockIdx.x, t = threadIdx.x;
    const float* s = scores + (size_t)row * kN + t * 16;
    float vals[16];
    float mx = -3.4e38f;
    #pragma unroll
    for (int i = 0; i < 16; ++i) { vals[i] = s[i]; mx = fmaxf(mx, vals[i]); }
    red[t] = mx; __syncthreads();
    for (int off = 128; off > 0; off >>= 1) {
        if (t < off) red[t] = fmaxf(red[t], red[t + off]);
        __syncthreads();
    }
    mx = red[0]; __syncthreads();
    float sum = 0.f;
    #pragma unroll
    for (int i = 0; i < 16; ++i) { vals[i] = __expf(vals[i] - mx); sum += vals[i]; }
    red[t] = sum; __syncthreads();
    for (int off = 128; off > 0; off >>= 1) {
        if (t < off) red[t] += red[t + off];
        __syncthreads();
    }
    float inv = 1.0f / red[0];
    v8bf o0, o1;
    #pragma unroll
    for (int i = 0; i < 8; ++i) {
        o0[i] = (__bf16)(vals[i] * inv);
        o1[i] = (__bf16)(vals[i + 8] * inv);
    }
    __bf16* p = probs + (size_t)row * kN + t * 16;
    *(v8bf*)p = o0;
    *((v8bf*)p + 1) = o1;
}

// ---------------------------------------------------------------------------
// 8) out = P @ V per (b,h): [16,4096] x [4096,64] -> fp32 [b][e][h*64+d].
//    One block per bh; 8 waves split K=4096 (512 each), LDS reduction.
// ---------------------------------------------------------------------------
__global__ void pv_wmma_kernel(const __bf16* __restrict__ probs,
                               const __bf16* __restrict__ vt,
                               float* __restrict__ attn_out) {
    __shared__ float part[8 * 4 * 8 * 32];  // 32 KB
    int bh = blockIdx.x;
    int t = threadIdx.x, lane = t & 31, wave = t >> 5;
    int khalf = (lane >> 4) << 4;
    int nbase = wave * 512;
    v8f acc[4] = {};
    for (int ks = 0; ks < 512; ks += 32) {
        int n0 = nbase + ks;
        v16bf a = *(const v16bf*)(probs + ((size_t)bh * kE + (lane & 15)) * kN + n0 + khalf);
        #pragma unroll
        for (int td = 0; td < 4; ++td) {
            int dd = td * 16 + (lane & 15);
            v16bf bf = *(const v16bf*)(vt + ((size_t)bh * kHD + dd) * kN + n0 + khalf);
            acc[td] = __builtin_amdgcn_wmma_f32_16x16x32_bf16(
                false, a, false, bf, (short)0, acc[td], false, false);
        }
    }
    #pragma unroll
    for (int td = 0; td < 4; ++td)
        #pragma unroll
        for (int r = 0; r < 8; ++r)
            part[((wave * 4 + td) * 8 + r) * 32 + lane] = acc[td][r];
    __syncthreads();
    int b = bh >> 3, h = bh & 7;
    for (int idx = t; idx < 1024; idx += 256) {
        int lanei = idx & 31;
        int r  = (idx >> 5) & 7;
        int td = idx >> 8;
        float s = 0.f;
        #pragma unroll
        for (int w = 0; w < 8; ++w) s += part[((w * 4 + td) * 8 + r) * 32 + lanei];
        int e  = r + ((lanei >> 4) << 3);
        int dd = td * 16 + (lanei & 15);
        attn_out[((size_t)b * kE + e) * kHID + h * kHD + dd] = s;
    }
}

// ---------------------------------------------------------------------------
// 9) fused epilogue per (b,e) row: proj + residual, RMSNorm, FFN (gelu), +res.
//    attn row staged into LDS with the CDNA5 async global->LDS path.
// ---------------------------------------------------------------------------
__global__ void epilogue_kernel(const float* __restrict__ attn_out,
                                const float* __restrict__ wproj, const float* __restrict__ bproj,
                                const float* __restrict__ mol_norm,
                                const float* __restrict__ ffn_norm_w,
                                const float* __restrict__ ffn_w1, const float* __restrict__ ffn_b1,
                                const float* __restrict__ ffn_w2, const float* __restrict__ ffn_b2,
                                float* __restrict__ out) {
    __shared__ __attribute__((aligned(16))) float xa[kM];
    __shared__ float att[kM];
    __shared__ float hn[kM];
    __shared__ float h1[4 * kM];
    __shared__ float red[256];
    int row = blockIdx.x;  // b*16 + e
    int t = threadIdx.x;
    if (t < 128) {  // 128 x b128 async transfers = 2KB row
        const float* src = attn_out + (size_t)row * kHID + t * 4;
        __builtin_amdgcn_global_load_async_to_lds_b128(
            (i32x4*)src, (i32x4*)&xa[t * 4], 0, 0);
    }
    __builtin_amdgcn_s_wait_asynccnt(0);
    __syncthreads();
    #pragma unroll
    for (int rep = 0; rep < 2; ++rep) {
        int j = t + rep * 256;
        float acc = bproj[j];
        for (int i = 0; i < kHID; ++i) acc += xa[i] * wproj[(size_t)i * kM + j];
        att[j] = acc + mol_norm[(size_t)row * kM + j];
    }
    __syncthreads();
    red[t] = att[t] * att[t] + att[t + 256] * att[t + 256];
    __syncthreads();
    for (int off = 128; off > 0; off >>= 1) {
        if (t < off) red[t] += red[t + off];
        __syncthreads();
    }
    float r = rsqrtf(red[0] * (1.0f / kM) + kEPS);
    hn[t]       = att[t] * r * ffn_norm_w[t];
    hn[t + 256] = att[t + 256] * r * ffn_norm_w[t + 256];
    __syncthreads();
    #pragma unroll
    for (int rep = 0; rep < 8; ++rep) {
        int j = t + rep * 256;
        float acc = ffn_b1[j];
        for (int i = 0; i < kM; ++i) acc += hn[i] * ffn_w1[(size_t)i * (4 * kM) + j];
        h1[j] = gelu_erf(acc);
    }
    __syncthreads();
    #pragma unroll
    for (int rep = 0; rep < 2; ++rep) {
        int j = t + rep * 256;
        float acc = ffn_b2[j];
        for (int i = 0; i < 4 * kM; ++i) acc += h1[i] * ffn_w2[(size_t)i * kM + j];
        out[(size_t)row * kM + j] = acc + att[j];
    }
}

// ---------------------------------------------------------------------------
// Launch
// ---------------------------------------------------------------------------
extern "C" void kernel_launch(void* const* d_in, const int* in_sizes, int n_in,
                              void* d_out, int out_size, void* d_ws, size_t ws_size,
                              hipStream_t stream) {
    const float* patch        = (const float*)d_in[0];
    const float* mol          = (const float*)d_in[1];
    const float* pe_w1        = (const float*)d_in[2];
    const float* pe_b1        = (const float*)d_in[3];
    const float* pe_w2        = (const float*)d_in[4];
    const float* pe_b2        = (const float*)d_in[5];
    const float* wq           = (const float*)d_in[6];
    const float* bq           = (const float*)d_in[7];
    const float* wkv          = (const float*)d_in[8];
    const float* bkv          = (const float*)d_in[9];
    const float* wproj        = (const float*)d_in[10];
    const float* bproj        = (const float*)d_in[11];
    const float* norm_mol_w   = (const float*)d_in[12];
    const float* norm_patch_w = (const float*)d_in[13];
    const float* ffn_w1       = (const float*)d_in[14];
    const float* ffn_b1       = (const float*)d_in[15];
    const float* ffn_w2       = (const float*)d_in[16];
    const float* ffn_b2       = (const float*)d_in[17];
    const float* ffn_norm_w   = (const float*)d_in[18];
    float* out = (float*)d_out;

    char* ws = (char*)d_ws;
    size_t off = 0;
    auto take = [&](size_t bytes) -> char* {
        char* p = ws + off;
        off += (bytes + 255) & ~(size_t)255;
        return p;
    };
    float*  pos_enc  = (float*) take((size_t)kN * kHID * 4);             // 8 MB
    __bf16* patchN   = (__bf16*)take((size_t)kB * kN * kC * 2);          // 16.8 MB
    __bf16* wkv_t    = (__bf16*)take((size_t)1024 * kC * 2);             // 0.5 MB
    __bf16* kmat     = (__bf16*)take((size_t)kB * kN * kHID * 2);        // 33.5 MB
    __bf16* vt       = (__bf16*)take((size_t)kB * kN * kHID * 2);        // 33.5 MB
    __bf16* qb       = (__bf16*)take((size_t)kB * kNH * kE * kHD * 2);   // 128 KB
    float*  mol_norm = (float*) take((size_t)kB * kE * kM * 4);          // 256 KB
    float*  scores   = (float*) take((size_t)kB * kNH * kE * kN * 4);    // 16.8 MB
    __bf16* probs    = (__bf16*)take((size_t)kB * kNH * kE * kN * 2);    // 8.4 MB
    float*  attn_out = (float*) take((size_t)kB * kE * kHID * 4);        // 256 KB

    pos_enc_kernel<<<kN, 256, 0, stream>>>(pe_w1, pe_b1, pe_w2, pe_b2, pos_enc);
    wkv_convert_kernel<<<1024, 256, 0, stream>>>(wkv, wkv_t);
    patch_norm_kernel<<<kB * (kN / 256), 256, 0, stream>>>(patch, norm_patch_w, patchN);
    kv_wmma_kernel<<<dim3(kB * kN / 16, 2), 256, 0, stream>>>(patchN, wkv_t, bkv, pos_enc, kmat, vt);
    q_mol_kernel<<<kB, 256, 0, stream>>>(mol, norm_mol_w, wq, bq, mol_norm, qb);
    scores_wmma_kernel<<<dim3(kB * kNH, kN / 512), 256, 0, stream>>>(qb, kmat, scores);
    softmax_kernel<<<kB * kNH * kE, 256, 0, stream>>>(scores, probs);
    pv_wmma_kernel<<<kB * kNH, 256, 0, stream>>>(probs, vt, attn_out);
    epilogue_kernel<<<kB * kE, 256, 0, stream>>>(attn_out, wproj, bproj, mol_norm,
                                                 ffn_norm_w, ffn_w1, ffn_b1, ffn_w2, ffn_b2, out);
}
